// GATv2DisplacerNet_26242250178985
// MI455X (gfx1250) — matrix operand order repaired and checked by reading, not compiled, not run on hardware
//
#include <hip/hip_runtime.h>

// ---------------------------------------------------------------------------
// CDNA5 / gfx1250 WMMA types
// ---------------------------------------------------------------------------
typedef __attribute__((ext_vector_type(16))) _Float16 v16h;
typedef __attribute__((ext_vector_type(8)))  float    v8f;

#define NPTS 16384
#define HCAT 992

// A-fragment k index for 16x16x32 f16 WMMA (ISA 7.12.2, 16-bit A 16x32 table):
//   lane group g = lane>>4, half i in [0,16): k = 8g + i + (i>=8 ? 8 : 0)
__device__ __forceinline__ int a_k(int g, int i) { return 8 * g + i + ((i & 8) ? 8 : 0); }

// ---------------------------------------------------------------------------
// seed: copy x [N,32] into concat buffer cols [0,32) (f32 + f16 mirror)
// ---------------------------------------------------------------------------
__global__ __launch_bounds__(256) void seed_x_kernel(const float* __restrict__ x,
                                                     float* __restrict__ hcat,
                                                     _Float16* __restrict__ hcath, int N) {
  int n = blockIdx.x * 8 + (threadIdx.x >> 5);
  int lane = threadIdx.x & 31;
  if (n >= N) return;
  float v = x[(size_t)n * 32 + lane];
  hcat[(size_t)n * HCAT + lane]  = v;
  hcath[(size_t)n * HCAT + lane] = (_Float16)v;
}

// ---------------------------------------------------------------------------
// row squared norms of a column segment of the concat buffer
// ---------------------------------------------------------------------------
__global__ __launch_bounds__(256) void sqnorm_kernel(const float* __restrict__ H, int ldh,
                                                     int colOff, int F,
                                                     float* __restrict__ sq, int N) {
  int n = blockIdx.x * blockDim.x + threadIdx.x;
  if (n >= N) return;
  const float* p = H + (size_t)n * ldh + colOff;
  float s = 0.f;
  for (int i = 0; i < F; ++i) s += p[i] * p[i];
  sq[n] = s;
}

// ---------------------------------------------------------------------------
// f32 -> f16 convert WITH transpose: in row-major [Kd][Nd] -> out [Nd][Kd].
// One-shot, tiny (weights <= 992x512); coalesced reads, scattered writes.
// ---------------------------------------------------------------------------
__global__ __launch_bounds__(256) void cvt_h_t_kernel(const float* __restrict__ in,
                                                      _Float16* __restrict__ out,
                                                      int Kd, int Nd) {
  int i = blockIdx.x * 256 + threadIdx.x;
  if (i >= Kd * Nd) return;
  int k = i / Nd, n = i % Nd;
  out[(size_t)n * Kd + k] = (_Float16)in[i];
}

// ---------------------------------------------------------------------------
// Fused KNN, templated on feature dim F. Per block: 128 query rows (8 waves
// x 16). Candidate tiles of 16 stream through LDS; each wave accumulates a
// 16x16 Gram tile with v_wmma_f32_16x16x32_f16; lanes 0..15 keep a register
// top-K=16 with dist = ||c||^2 - 2 q.c (||q||^2 drops out of the ranking).
// Cs rows padded (+8 halves) -> conflict-free, 16B-aligned ds_load_b128.
// ---------------------------------------------------------------------------
template <int F>
__global__ __launch_bounds__(256) void knn_kernel(const _Float16* __restrict__ H, int ldh,
                                                  int colOff,
                                                  const float* __restrict__ sq,
                                                  int* __restrict__ knn, int N) {
  constexpr int NCH = F / 32;        // K-chunks of 32 (F in {32,64,128,256})
  constexpr int PER = F / 16;        // halves per staging thread
  constexpr int LDP = F + 8;         // padded LDS row (bank-conflict free)
  __shared__ _Float16 Cs[16][LDP];   // candidate tile, candidate-major
  __shared__ float    sqs[16];
  __shared__ float    dts[8][16][16];

  int tid = threadIdx.x, wave = tid >> 5, lane = tid & 31;
  int lm = lane & 15, lg = lane >> 4;
  int q0 = blockIdx.x * 128 + wave * 16;

  // load this wave's A fragments once (query rows, f16; contiguous 16-half runs, k = 16*lg + i)
  v16h af[NCH];
#pragma unroll
  for (int c = 0; c < NCH; ++c) {
    const _Float16* src = H + (size_t)(q0 + lm) * ldh + colOff + c * 32;
#pragma unroll
    for (int i = 0; i < 16; ++i) af[c][i] = src[lg * 16 + i];
  }

  float best[16]; int bidx[16];
#pragma unroll
  for (int j = 0; j < 16; ++j) { best[j] = 3.4e38f; bidx[j] = 0; }
  float worst = 3.4e38f;

  for (int c0 = 0; c0 < N; c0 += 16) {
    { // cooperative stage of 16 candidate rows x F halves
      int r = tid >> 4, seg = tid & 15;
      const _Float16* src = H + (size_t)(c0 + r) * ldh + colOff + seg * PER;
#pragma unroll
      for (int i = 0; i < PER; ++i) Cs[r][seg * PER + i] = src[i];
      if (c0 + 16 < N)
        __builtin_prefetch(H + (size_t)(c0 + 16 + r) * ldh + colOff + seg * PER, 0, 1);
      if (tid < 16) sqs[tid] = sq[c0 + tid];
    }
    __syncthreads();

    v8f acc = {};
#pragma unroll
    for (int c = 0; c < NCH; ++c) {
      v16h bf;
#pragma unroll
      for (int i = 0; i < 16; ++i) bf[i] = Cs[lm][c * 32 + lg * 16 + i];
      acc = __builtin_amdgcn_wmma_f32_16x16x32_f16(false, af[c], false, bf,
                                                   (short)0, acc, false, false);
    }

    // C layout: element (m = j + 8*lg, n = lm)
#pragma unroll
    for (int j = 0; j < 8; ++j) {
      int m = 8 * lg + j;
      float d = sqs[lm] - 2.0f * acc[j];
      if (c0 + lm == q0 + m) d = 3.4e38f;   // exclude self
      dts[wave][m][lm] = d;
    }
    __syncthreads();

    if (lane < 16) {                 // one lane per query row: top-K upkeep
#pragma unroll
      for (int n = 0; n < 16; ++n) {
        float d = dts[wave][lane][n];
        if (d < worst) {
          int slot = 0; float mx = best[0];
#pragma unroll
          for (int j = 1; j < 16; ++j) if (best[j] > mx) { mx = best[j]; slot = j; }
#pragma unroll
          for (int j = 0; j < 16; ++j) if (j == slot) { best[j] = d; bidx[j] = c0 + n; }
          mx = best[0];
#pragma unroll
          for (int j = 1; j < 16; ++j) mx = fmaxf(mx, best[j]);
          worst = mx;
        }
      }
    }
    __syncthreads();
  }
  if (lane < 16) {
    int q = q0 + lane;
#pragma unroll
    for (int j = 0; j < 16; ++j) knn[(size_t)q * 16 + j] = bidx[j];
  }
}

// ---------------------------------------------------------------------------
// WMMA GEMM with pre-transposed B:
//   C[M,Nd] = A[M,Kd](f16,row-major,lda) x B[Kd,Nd], B given as Bt[Nd][Kd].
// Block = 256 threads (8 waves): 128(M) x 64(N) tile; K in chunks of 32.
// LDS tiles stored n-major with padded rows (40 halves = 80B: 16B-aligned,
// conflict-free) so every fragment read is two ds_load_b128.
// Epilogue: optional bias, relu, f32 and/or f16 outputs.
// ---------------------------------------------------------------------------
__global__ __launch_bounds__(256) void gemm_f16_kernel(const _Float16* __restrict__ A, int lda,
                                                       const _Float16* __restrict__ Bt,
                                                       float* __restrict__ Cf, int ldcf,
                                                       _Float16* __restrict__ Ch, int ldch,
                                                       const float* __restrict__ bias,
                                                       int M, int Kd, int Nd, int act) {
  __shared__ _Float16 As[128][40];
  __shared__ _Float16 Bts[64][40];
  int tid = threadIdx.x, wave = tid >> 5, lane = tid & 31;
  int lm = lane & 15, lg = lane >> 4;
  int bm = blockIdx.x * 128, bn = blockIdx.y * 64;

  v8f acc[4];
  { v8f z = {}; acc[0] = z; acc[1] = z; acc[2] = z; acc[3] = z; }

  for (int k0 = 0; k0 < Kd; k0 += 32) {
    { // A: 128x32 halves, 16 contiguous per thread
      int r = tid >> 1, c = (tid & 1) * 16;
      const _Float16* src = A + (size_t)(bm + r) * lda + k0 + c;
#pragma unroll
      for (int i = 0; i < 16; ++i) As[r][c + i] = src[i];
      if (k0 + 32 < Kd) __builtin_prefetch(src + 32, 0, 1);
    }
    { // Bt: 64 rows (n) x 32 halves (k), 8 contiguous per thread
      int r = tid >> 2, seg = (tid & 3) * 8;
      if (bn + r < Nd) {
        const _Float16* src = Bt + (size_t)(bn + r) * Kd + k0 + seg;
#pragma unroll
        for (int i = 0; i < 8; ++i) Bts[r][seg + i] = src[i];
      } else {
#pragma unroll
        for (int i = 0; i < 8; ++i) Bts[r][seg + i] = (_Float16)0.f;
      }
    }
    __syncthreads();

    v16h afr;
    int mrow = wave * 16 + lm;
#pragma unroll
    for (int i = 0; i < 16; ++i) afr[i] = As[mrow][lg * 16 + i];
#pragma unroll
    for (int t = 0; t < 4; ++t) {
      v16h bfr;   // B[k][n] with k = 16*lg + i, n = bn + 16t + lm  (contiguous i)
#pragma unroll
      for (int i = 0; i < 16; ++i) bfr[i] = Bts[t * 16 + lm][lg * 16 + i];
      acc[t] = __builtin_amdgcn_wmma_f32_16x16x32_f16(false, afr, false, bfr,
                                                      (short)0, acc[t], false, false);
    }
    __syncthreads();
  }

  float bv[4];
#pragma unroll
  for (int t = 0; t < 4; ++t) {
    int n = bn + t * 16 + lm;
    bv[t] = (bias && n < Nd) ? bias[n] : 0.f;
  }
#pragma unroll
  for (int t = 0; t < 4; ++t) {
    int n = bn + t * 16 + lm;
    if (n < Nd) {
#pragma unroll
      for (int j = 0; j < 8; ++j) {
        int m = bm + wave * 16 + 8 * lg + j;
        float v = acc[t][j] + bv[t];
        if (act) v = fmaxf(v, 0.f);
        if (Cf) Cf[(size_t)m * ldcf + n] = v;
        if (Ch) Ch[(size_t)m * ldch + n] = (_Float16)v;
      }
    }
  }
}

// ---------------------------------------------------------------------------
// GATv2 attention + aggregate, templated on F: one wave per node.
// e_j = att . lrelu(xl_nb + xr), softmax over K=16, out = sum alpha_j xl_nb + b
// ---------------------------------------------------------------------------
template <int F>
__global__ __launch_bounds__(256) void gat_attn_kernel(const float* __restrict__ xl,
                                                       const float* __restrict__ xr,
                                                       const int* __restrict__ knn,
                                                       const float* __restrict__ att,
                                                       const float* __restrict__ bias,
                                                       float* __restrict__ outF, int ldof,
                                                       _Float16* __restrict__ outH, int ldoh,
                                                       int colOff, int N) {
  constexpr int R = F / 32;  // 2..16 per-lane elements
  int wave = threadIdx.x >> 5, lane = threadIdx.x & 31;
  int n = blockIdx.x * 8 + wave;
  if (n >= N) return;

  float xrv[R], atv[R], accv[R];
#pragma unroll
  for (int i = 0; i < R; ++i) {
    xrv[i]  = xr[(size_t)n * F + i * 32 + lane];
    atv[i]  = att[i * 32 + lane];
    accv[i] = 0.f;
  }
  int nb[16];
#pragma unroll
  for (int j = 0; j < 16; ++j) nb[j] = knn[(size_t)n * 16 + j];

  float e[16];
#pragma unroll
  for (int j = 0; j < 16; ++j) {
    const float* xlp = xl + (size_t)nb[j] * F;
    float s = 0.f;
#pragma unroll
    for (int i = 0; i < R; ++i) {
      float t = xlp[i * 32 + lane] + xrv[i];
      t = (t > 0.f) ? t : 0.2f * t;   // leaky relu, slope 0.2
      s += t * atv[i];
    }
#pragma unroll
    for (int m = 16; m >= 1; m >>= 1) s += __shfl_xor(s, m, 32);
    e[j] = s;
  }
  float mx = e[0];
#pragma unroll
  for (int j = 1; j < 16; ++j) mx = fmaxf(mx, e[j]);
  float den = 0.f;
#pragma unroll
  for (int j = 0; j < 16; ++j) { e[j] = __expf(e[j] - mx); den += e[j]; }
  float inv = 1.f / den;
#pragma unroll
  for (int j = 0; j < 16; ++j) {
    float a = e[j] * inv;
    const float* xlp = xl + (size_t)nb[j] * F;
#pragma unroll
    for (int i = 0; i < R; ++i) accv[i] += a * xlp[i * 32 + lane];
  }
#pragma unroll
  for (int i = 0; i < R; ++i) {
    int col = i * 32 + lane;
    float v = accv[i] + bias[col];
    outF[(size_t)n * ldof + colOff + col] = v;
    outH[(size_t)n * ldoh + colOff + col] = (_Float16)v;
  }
}

// ---------------------------------------------------------------------------
// host: carve scratch, run 4 GATv2 layers + 3-layer MLP
// ---------------------------------------------------------------------------
extern "C" void kernel_launch(void* const* d_in, const int* in_sizes, int n_in,
                              void* d_out, int out_size, void* d_ws, size_t ws_size,
                              hipStream_t stream) {
  (void)in_sizes; (void)n_in; (void)out_size; (void)ws_size;
  const int N = NPTS;
  const float* x = (const float*)d_in[0];

  char* p = (char*)d_ws;
  auto carve = [&](size_t bytes) -> void* {
    void* r = (void*)p; p += (bytes + 255) & ~(size_t)255; return r;
  };
  float*    hcat  = (float*)   carve((size_t)N * HCAT * 4);
  _Float16* hcath = (_Float16*)carve((size_t)N * HCAT * 2);
  float*    xl    = (float*)   carve((size_t)N * 512 * 4);
  float*    xr    = (float*)   carve((size_t)N * 512 * 4);
  int*      knn   = (int*)     carve((size_t)N * 16 * 4);
  float*    sq    = (float*)   carve((size_t)N * 4);
  _Float16* wh0   = (_Float16*)carve((size_t)992 * 512 * 2);
  _Float16* wh1   = (_Float16*)carve((size_t)992 * 512 * 2);
  _Float16* m1h   = (_Float16*)carve((size_t)N * 512 * 2);
  _Float16* m2h   = (_Float16*)carve((size_t)N * 256 * 2);

  seed_x_kernel<<<N / 8, 256, 0, stream>>>(x, hcat, hcath, N);

  const int off[5]  = {0, 32, 96, 224, 480};
  const int fin[4]  = {32, 64, 128, 256};
  const int fout[4] = {64, 128, 256, 512};

  for (int l = 0; l < 4; ++l) {
    const float* Wl  = (const float*)d_in[1 + 4 * l];
    const float* Wr  = (const float*)d_in[2 + 4 * l];
    const float* att = (const float*)d_in[3 + 4 * l];
    const float* b   = (const float*)d_in[4 + 4 * l];

    sqnorm_kernel<<<(N + 255) / 256, 256, 0, stream>>>(hcat, HCAT, off[l], fin[l], sq, N);

    switch (fin[l]) {   // compile-time specialized fused KNN
      case 32:  knn_kernel<32 ><<<N / 128, 256, 0, stream>>>(hcath, HCAT, off[l], sq, knn, N); break;
      case 64:  knn_kernel<64 ><<<N / 128, 256, 0, stream>>>(hcath, HCAT, off[l], sq, knn, N); break;
      case 128: knn_kernel<128><<<N / 128, 256, 0, stream>>>(hcath, HCAT, off[l], sq, knn, N); break;
      default:  knn_kernel<256><<<N / 128, 256, 0, stream>>>(hcath, HCAT, off[l], sq, knn, N); break;
    }

    int wn = fin[l] * fout[l];
    cvt_h_t_kernel<<<(wn + 255) / 256, 256, 0, stream>>>(Wl, wh0, fin[l], fout[l]);
    cvt_h_t_kernel<<<(wn + 255) / 256, 256, 0, stream>>>(Wr, wh1, fin[l], fout[l]);

    dim3 gg(N / 128, (fout[l] + 63) / 64);
    gemm_f16_kernel<<<gg, 256, 0, stream>>>(hcath + off[l], HCAT, wh0,
                                            xl, fout[l], nullptr, 0, nullptr,
                                            N, fin[l], fout[l], 0);
    gemm_f16_kernel<<<gg, 256, 0, stream>>>(hcath + off[l], HCAT, wh1,
                                            xr, fout[l], nullptr, 0, nullptr,
                                            N, fin[l], fout[l], 0);

    switch (fout[l]) {  // compile-time specialized attention
      case 64:  gat_attn_kernel<64 ><<<N / 8, 256, 0, stream>>>(xl, xr, knn, att, b, hcat, HCAT, hcath, HCAT, off[l + 1], N); break;
      case 128: gat_attn_kernel<128><<<N / 8, 256, 0, stream>>>(xl, xr, knn, att, b, hcat, HCAT, hcath, HCAT, off[l + 1], N); break;
      case 256: gat_attn_kernel<256><<<N / 8, 256, 0, stream>>>(xl, xr, knn, att, b, hcat, HCAT, hcath, HCAT, off[l + 1], N); break;
      default:  gat_attn_kernel<512><<<N / 8, 256, 0, stream>>>(xl, xr, knn, att, b, hcat, HCAT, hcath, HCAT, off[l + 1], N); break;
    }
  }

  // MLP: 992 -> 512 (relu) -> 256 (relu) -> 3
  {
    int wn = 992 * 512;
    cvt_h_t_kernel<<<(wn + 255) / 256, 256, 0, stream>>>((const float*)d_in[17], wh0, 992, 512);
    dim3 g1(N / 128, 512 / 64);
    gemm_f16_kernel<<<g1, 256, 0, stream>>>(hcath, HCAT, wh0,
                                            nullptr, 0, m1h, 512, (const float*)d_in[18],
                                            N, 992, 512, 1);
    wn = 512 * 256;
    cvt_h_t_kernel<<<(wn + 255) / 256, 256, 0, stream>>>((const float*)d_in[19], wh0, 512, 256);
    dim3 g2(N / 128, 256 / 64);
    gemm_f16_kernel<<<g2, 256, 0, stream>>>(m1h, 512, wh0,
                                            nullptr, 0, m2h, 256, (const float*)d_in[20],
                                            N, 512, 256, 1);
    wn = 256 * 3;
    cvt_h_t_kernel<<<(wn + 255) / 256, 256, 0, stream>>>((const float*)d_in[21], wh0, 256, 3);
    dim3 g3(N / 128, 1);
    gemm_f16_kernel<<<g3, 256, 0, stream>>>(m2h, 256, wh0,
                                            (float*)d_out, 3, nullptr, 0, (const float*)d_in[22],
                                            N, 256, 3, 0);
  }
}
